// MultiHeadedAttention_76158360093308
// MI455X (gfx1250) — matrix operand order repaired and checked by reading, not compiled, not run on hardware
//
#include <hip/hip_runtime.h>
#include <stdint.h>

// ---------------- problem constants ----------------
#define BB 2
#define SS 2048
#define DD 1024
#define HH 16
#define HD 64
#define MM (BB * SS)          // 4096 rows for projection GEMMs

typedef __attribute__((ext_vector_type(16))) __bf16 v16bf;
typedef __attribute__((ext_vector_type(8)))  float  v8f;
typedef __attribute__((ext_vector_type(4)))  float  f32x4;

union FragB16 {
    uint4    q[2];
    uint32_t d[8];
    v16bf    v;
};

static __device__ __forceinline__ v8f wmma_bf16(v16bf a, v16bf b, v8f c) {
    return __builtin_amdgcn_wmma_f32_16x16x32_bf16(
        /*neg_a=*/false, a, /*neg_b=*/false, b,
        /*c_mod=*/(short)0, c, /*reuse_a=*/false, /*reuse_b=*/false);
}

static __device__ __forceinline__ uint16_t f32_to_bf16(float f) {
    uint32_t u = __float_as_uint(f);
    u = (u + 0x7FFFu + ((u >> 16) & 1u)) >> 16;   // round to nearest even
    return (uint16_t)u;
}

static __device__ __forceinline__ uint32_t pack_bf16x2(float lo, float hi) {
    uint32_t a = __float_as_uint(lo);
    uint32_t b = __float_as_uint(hi);
    a = (a + 0x7FFFu + ((a >> 16) & 1u)) >> 16;
    b = (b + 0x7FFFu + ((b >> 16) & 1u)) & 0xFFFF0000u;
    return (a & 0xFFFFu) | b;
}

// async copy of 16B global -> LDS (per-lane addresses), tracked by ASYNCcnt
static __device__ __forceinline__ void async_cp16(uint32_t lds_addr, const void* gptr) {
    asm volatile("global_load_async_to_lds_b128 %0, %1, off"
                 :: "v"(lds_addr), "v"((uint64_t)(uintptr_t)gptr)
                 : "memory");
}
static __device__ __forceinline__ void wait_async0() {
    asm volatile("s_wait_asynccnt 0x0" ::: "memory");
}

// ---------------- f32 -> bf16 conversion (2 elems/thread, b32 stores) ----------------
__global__ void cvt_f32_bf16_kernel(const float* __restrict__ in,
                                    uint16_t* __restrict__ out, int n2) {
    int i = blockIdx.x * blockDim.x + threadIdx.x;
    if (i < n2) {
        float2 f = reinterpret_cast<const float2*>(in)[i];
        reinterpret_cast<uint32_t*>(out)[i] = pack_bf16x2(f.x, f.y);
    }
}

// ---------------- projection GEMM: OUT = X[bf16 MxK] @ W^T[bf16 NxK] + bias ----------------
// W tile (128 cols x 32 k, 8KB) is staged into LDS with async-to-LDS copies,
// double-buffered (one barrier per K-step); A fragments load direct from global.
// mode 0: bf16 row-major [M,N], value scaled          (Q with 1/8, K with 1.0)
// mode 1: bf16 transposed per-head -> Vt[b,h][d][s]   (V)
// mode 2: f32 row-major [M,N]                          (final output)
__global__ __launch_bounds__(256) void gemm_proj_kernel(
    const uint16_t* __restrict__ X, const uint16_t* __restrict__ W,
    const float* __restrict__ bias, void* __restrict__ outp,
    int mode, float scale) {
    __shared__ uint16_t ldsW[2][128 * 32];   // 2 x 8KB double buffer

    const int lane = threadIdx.x & 31;
    const int wave = threadIdx.x >> 5;
    const int wm = wave >> 1;             // 0..3
    const int wn = wave & 1;              // 0..1
    const int lm = lane & 15;
    const int half = lane >> 4;
    const int m0 = blockIdx.x * 128 + wm * 32;
    const int nblk = blockIdx.y * 128;
    const int n0 = nblk + wn * 64;

    // issue the async copies for one 128x32 W tile into buffer `buf`
    auto issueB = [&](int kk, int buf) {
#pragma unroll
        for (int c = 0; c < 2; ++c) {
            const int chunk = threadIdx.x * 2 + c;   // 0..511
            const int rowN = chunk >> 2;             // 0..127
            const int part = chunk & 3;              // 4 x 8 elems per row
            const uint16_t* g = W + (size_t)(nblk + rowN) * DD + kk + part * 8;
            const uint32_t l =
                (uint32_t)(uintptr_t)&ldsW[buf][rowN * 32 + part * 8];
            async_cp16(l, g);
        }
    };

    v8f acc[2][4] = {};

    issueB(0, 0);
    for (int k = 0; k < DD; k += 32) {
        const int cur = (k >> 5) & 1;
        wait_async0();          // this wave's copies for buf[cur] have landed
        __syncthreads();        // everyone's copies landed; prior readers done
        if (k + 32 < DD) issueB(k + 32, cur ^ 1);

        FragB16 a[2], b[4];
#pragma unroll
        for (int i = 0; i < 2; ++i) {
            const uint16_t* p = X + (size_t)(m0 + i * 16 + lm) * DD + k + half * 8;
            a[i].q[0] = *reinterpret_cast<const uint4*>(p);
            a[i].q[1] = *reinterpret_cast<const uint4*>(p + 16);
        }
#pragma unroll
        for (int j = 0; j < 4; ++j) {
            const int nl = wn * 64 + j * 16 + lm;    // column within the 128-wide tile
            const uint4* lp =
                reinterpret_cast<const uint4*>(&ldsW[cur][nl * 32 + half * 8]);
            b[j].q[0] = lp[0];
            b[j].q[1] = lp[2];
        }
#pragma unroll
        for (int i = 0; i < 2; ++i)
#pragma unroll
            for (int j = 0; j < 4; ++j)
                acc[i][j] = wmma_bf16(a[i].v, b[j].v, acc[i][j]);
    }

#pragma unroll
    for (int j = 0; j < 4; ++j) {
        const int col = n0 + j * 16 + lm;
        const float bv = bias[col];
#pragma unroll
        for (int i = 0; i < 2; ++i) {
#pragma unroll
            for (int r = 0; r < 8; ++r) {
                const int row = m0 + i * 16 + r + half * 8;
                const float val = (acc[i][j][r] + bv) * scale;
                if (mode == 0) {
                    reinterpret_cast<uint16_t*>(outp)[(size_t)row * DD + col] = f32_to_bf16(val);
                } else if (mode == 1) {
                    const int bb = row >> 11;           // batch
                    const int ss = row & (SS - 1);      // sequence pos
                    reinterpret_cast<uint16_t*>(outp)[((size_t)(bb * DD + col)) * SS + ss] =
                        f32_to_bf16(val);
                } else {
                    reinterpret_cast<float*>(outp)[(size_t)row * DD + col] = val;
                }
            }
        }
    }
}

// ---------------- scores = Q @ K^T (causal lower-triangle tiles only) ----------------
// grid: (136 tile-pairs, H, B); workgroup tile 128x128, wave tile 32x64
// raw scores go out with non-temporal stores (537MB stream, keep L2 for Q/K/Vt)
__global__ __launch_bounds__(256) void attn_scores_kernel(
    const uint16_t* __restrict__ Q, const uint16_t* __restrict__ Km,
    float* __restrict__ attn) {
    const int t = blockIdx.x;
    int qt = 0;
    while ((qt + 1) * (qt + 2) / 2 <= t) ++qt;
    const int kt = t - qt * (qt + 1) / 2;
    const int h = blockIdx.y;
    const int bidx = blockIdx.z;

    const int lane = threadIdx.x & 31;
    const int wave = threadIdx.x >> 5;
    const int wm = wave >> 1, wn = wave & 1;
    const int lm = lane & 15, half = lane >> 4;
    const int q0 = qt * 128 + wm * 32;
    const int c0 = kt * 128 + wn * 64;

    v8f acc[2][4] = {};

    for (int d = 0; d < HD; d += 32) {
        FragB16 a[2], b[4];
#pragma unroll
        for (int i = 0; i < 2; ++i) {
            const uint16_t* p = Q + (size_t)(bidx * SS + q0 + i * 16 + lm) * DD
                                  + h * HD + d + half * 8;
            a[i].q[0] = *reinterpret_cast<const uint4*>(p);
            a[i].q[1] = *reinterpret_cast<const uint4*>(p + 16);
        }
#pragma unroll
        for (int j = 0; j < 4; ++j) {
            const uint16_t* p = Km + (size_t)(bidx * SS + c0 + j * 16 + lm) * DD
                                   + h * HD + d + half * 8;
            b[j].q[0] = *reinterpret_cast<const uint4*>(p);
            b[j].q[1] = *reinterpret_cast<const uint4*>(p + 16);
        }
#pragma unroll
        for (int i = 0; i < 2; ++i)
#pragma unroll
            for (int j = 0; j < 4; ++j)
                acc[i][j] = wmma_bf16(a[i].v, b[j].v, acc[i][j]);
    }

    const size_t base = ((size_t)(bidx * HH + h)) * SS * SS;
#pragma unroll
    for (int j = 0; j < 4; ++j) {
        const int col = c0 + j * 16 + lm;
#pragma unroll
        for (int i = 0; i < 2; ++i) {
#pragma unroll
            for (int r = 0; r < 8; ++r) {
                const int row = q0 + i * 16 + r + half * 8;
                __builtin_nontemporal_store(acc[i][j][r],
                                            &attn[base + (size_t)row * SS + col]);
            }
        }
    }
}

// ---------------- row softmax with causal mask, in-place on attn ----------------
// one wave32 per row; full 2048-float row kept in 64 VGPRs per lane.
// NT loads/stores: this is a pure 2x537MB HBM stream.
__global__ __launch_bounds__(256) void softmax_kernel(float* __restrict__ attn) {
    const int wave = threadIdx.x >> 5;
    const int lane = threadIdx.x & 31;
    const size_t row = (size_t)blockIdx.x * 8 + wave;   // (b*H+h)*S + q
    const int q = (int)(row & (SS - 1));
    float* base = attn + row * SS;
    const int nv = q + 1;

    float v[64];
#pragma unroll
    for (int c = 0; c < 16; ++c) {
        f32x4 t = __builtin_nontemporal_load(
            reinterpret_cast<const f32x4*>(base + c * 128 + lane * 4));
        *reinterpret_cast<f32x4*>(&v[c * 4]) = t;
    }

    float m = -3.0e38f;
#pragma unroll
    for (int c = 0; c < 16; ++c) {
        const int kb = c * 128 + lane * 4;
#pragma unroll
        for (int j = 0; j < 4; ++j)
            if (kb + j < nv) m = fmaxf(m, v[c * 4 + j]);
    }
#pragma unroll
    for (int off = 16; off >= 1; off >>= 1)
        m = fmaxf(m, __shfl_xor(m, off, 32));

    float s = 0.f;
#pragma unroll
    for (int c = 0; c < 16; ++c) {
        const int kb = c * 128 + lane * 4;
#pragma unroll
        for (int j = 0; j < 4; ++j) {
            const int i = c * 4 + j;
            if (kb + j < nv) {
                const float e = __expf(v[i] - m);
                v[i] = e;
                s += e;
            } else {
                v[i] = 0.f;
            }
        }
    }
#pragma unroll
    for (int off = 16; off >= 1; off >>= 1)
        s += __shfl_xor(s, off, 32);
    const float inv = 1.0f / s;

#pragma unroll
    for (int i = 0; i < 64; ++i) v[i] *= inv;
#pragma unroll
    for (int c = 0; c < 16; ++c)
        __builtin_nontemporal_store(*reinterpret_cast<const f32x4*>(&v[c * 4]),
                                    reinterpret_cast<f32x4*>(base + c * 128 + lane * 4));
}

// ---------------- ctx = attn @ V  (per head; K-loop truncated by causal limit) ----------------
// grid: (16 q-tiles, H, B); workgroup tile 128x64, wave tile 32x32
// attn re-read is NT (one-shot stream); Vt stays cache-resident.
__global__ __launch_bounds__(256) void attn_ctx_kernel(
    const float* __restrict__ attn, const uint16_t* __restrict__ Vt,
    uint16_t* __restrict__ ctx) {
    const int qt = blockIdx.x;
    const int h = blockIdx.y;
    const int bidx = blockIdx.z;

    const int lane = threadIdx.x & 31;
    const int wave = threadIdx.x >> 5;
    const int wm = wave >> 1, wn = wave & 1;
    const int lm = lane & 15, half = lane >> 4;
    const int q0 = qt * 128 + wm * 32;
    const int d0 = wn * 32;

    v8f acc[2][2] = {};
    const int klim = (qt + 1) * 128;
    const size_t abase = ((size_t)(bidx * HH + h)) * SS * SS;

    for (int k = 0; k < klim; k += 32) {
        FragB16 a[2], b[2];
#pragma unroll
        for (int i = 0; i < 2; ++i) {
            const float* pr = attn + abase + (size_t)(q0 + i * 16 + lm) * SS + k;
            const f32x4 f0 = __builtin_nontemporal_load(
                reinterpret_cast<const f32x4*>(pr + half * 8));
            const f32x4 f1 = __builtin_nontemporal_load(
                reinterpret_cast<const f32x4*>(pr + half * 8 + 4));
            const f32x4 f2 = __builtin_nontemporal_load(
                reinterpret_cast<const f32x4*>(pr + 16 + half * 8));
            const f32x4 f3 = __builtin_nontemporal_load(
                reinterpret_cast<const f32x4*>(pr + 16 + half * 8 + 4));
            a[i].d[0] = pack_bf16x2(f0.x, f0.y);
            a[i].d[1] = pack_bf16x2(f0.z, f0.w);
            a[i].d[2] = pack_bf16x2(f1.x, f1.y);
            a[i].d[3] = pack_bf16x2(f1.z, f1.w);
            a[i].d[4] = pack_bf16x2(f2.x, f2.y);
            a[i].d[5] = pack_bf16x2(f2.z, f2.w);
            a[i].d[6] = pack_bf16x2(f3.x, f3.y);
            a[i].d[7] = pack_bf16x2(f3.z, f3.w);
        }
#pragma unroll
        for (int j = 0; j < 2; ++j) {
            const uint16_t* p = Vt + (size_t)(bidx * DD + h * HD + d0 + j * 16 + lm) * SS
                                   + k + half * 8;
            b[j].q[0] = *reinterpret_cast<const uint4*>(p);
            b[j].q[1] = *reinterpret_cast<const uint4*>(p + 16);
        }
#pragma unroll
        for (int i = 0; i < 2; ++i)
#pragma unroll
            for (int j = 0; j < 2; ++j)
                acc[i][j] = wmma_bf16(a[i].v, b[j].v, acc[i][j]);
    }

#pragma unroll
    for (int j = 0; j < 2; ++j) {
        const int dcol = d0 + j * 16 + lm;
#pragma unroll
        for (int i = 0; i < 2; ++i) {
#pragma unroll
            for (int r = 0; r < 8; ++r) {
                const int qrow = q0 + i * 16 + r + half * 8;
                ctx[((size_t)(bidx * SS + qrow)) * DD + h * HD + dcol] =
                    f32_to_bf16(acc[i][j][r]);
            }
        }
    }
}

// ---------------- host-side launch ----------------
extern "C" void kernel_launch(void* const* d_in, const int* in_sizes, int n_in,
                              void* d_out, int out_size, void* d_ws, size_t ws_size,
                              hipStream_t stream) {
    const float* query = (const float*)d_in[0];
    const float* key   = (const float*)d_in[1];
    const float* value = (const float*)d_in[2];
    // d_in[3] = mask: causal by construction; applied analytically in softmax
    const float* Wq = (const float*)d_in[4];
    const float* bq = (const float*)d_in[5];
    const float* Wk = (const float*)d_in[6];
    const float* bk = (const float*)d_in[7];
    const float* Wv = (const float*)d_in[8];
    const float* bv = (const float*)d_in[9];
    const float* Wo = (const float*)d_in[10];
    const float* bo = (const float*)d_in[11];

    float* out  = (float*)d_out;
    float* attn = out + (size_t)MM * DD;         // output first, then attn

    const size_t XSZ = (size_t)MM * DD;          // 4096*1024 bf16 elements
    const size_t WSZ = (size_t)DD * DD;          // 1024*1024 bf16 elements
    uint16_t* w   = (uint16_t*)d_ws;
    uint16_t* xq  = w;
    uint16_t* xk  = xq + XSZ;
    uint16_t* xv  = xk + XSZ;
    uint16_t* wqb = xv + XSZ;
    uint16_t* wkb = wqb + WSZ;
    uint16_t* wvb = wkb + WSZ;
    uint16_t* wob = wvb + WSZ;
    uint16_t* Qb  = wob + WSZ;
    uint16_t* Kb  = Qb + XSZ;
    uint16_t* Vt  = Kb + XSZ;
    uint16_t* Cx  = Vt + XSZ;

    auto cvt = [&](const float* src, uint16_t* dst, size_t n) {
        const int n2 = (int)(n / 2);
        cvt_f32_bf16_kernel<<<(n2 + 255) / 256, 256, 0, stream>>>(src, dst, n2);
    };
    cvt(query, xq, XSZ);
    cvt(key,   xk, XSZ);
    cvt(value, xv, XSZ);
    cvt(Wq, wqb, WSZ);
    cvt(Wk, wkb, WSZ);
    cvt(Wv, wvb, WSZ);
    cvt(Wo, wob, WSZ);

    const dim3 gproj(MM / 128, DD / 128);
    gemm_proj_kernel<<<gproj, 256, 0, stream>>>(xq, wqb, bq, Qb, 0, 0.125f); // Q / sqrt(HD)
    gemm_proj_kernel<<<gproj, 256, 0, stream>>>(xk, wkb, bk, Kb, 0, 1.0f);   // K
    gemm_proj_kernel<<<gproj, 256, 0, stream>>>(xv, wvb, bv, Vt, 1, 1.0f);   // V -> Vt

    attn_scores_kernel<<<dim3(136, HH, BB), 256, 0, stream>>>(Qb, Kb, attn);
    softmax_kernel<<<(BB * HH * SS) / 8, 256, 0, stream>>>(attn);
    attn_ctx_kernel<<<dim3(SS / 128, HH, BB), 256, 0, stream>>>(attn, Vt, Cx);

    gemm_proj_kernel<<<gproj, 256, 0, stream>>>(Cx, wob, bo, out, 2, 1.0f);  // output proj
}
